// OptimizedQuantizedExpert_43508018708617
// MI455X (gfx1250) — 8-bit matrix operand order repaired and checked by reading, compiled, not run on hardware
//
#include <hip/hip_runtime.h>
#include <cstdint>
#include <cstddef>


typedef __attribute__((ext_vector_type(8))) int   v8i;

#define T_TOK   4096
#define D_MODEL 4096
#define HIDDEN  10922
#define HPAD    11008
#define QMAXF   127.0f

#define LDPITCH 80

__device__ __forceinline__ void async_ld_b128(void* lds, const void* gptr) {
  unsigned lo = (unsigned)(uintptr_t)lds;
  asm volatile("global_load_async_to_lds_b128 %0, %1, off"
               :: "v"(lo), "v"(gptr) : "memory");
}

__device__ __forceinline__ float q_scale(const unsigned* slot) {
  float m = __uint_as_float(*slot);
  return fmaxf(m * (1.0f / QMAXF), 1e-8f);
}

__device__ __forceinline__ float fq_val(float v, float s) {
  float q = fminf(fmaxf(rintf(v / s), -QMAXF), QMAXF);
  return q * s;
}

__global__ void init_slots_kernel(unsigned* slots) {
  if (threadIdx.x < 8) slots[threadIdx.x] = 0u;
}

__global__ __launch_bounds__(256) void absmax_kernel(const float* __restrict__ src,
                                                     size_t n, unsigned* out) {
  float lmax = 0.0f;
  size_t stride = (size_t)gridDim.x * blockDim.x;
  for (size_t i = (size_t)blockIdx.x * blockDim.x + threadIdx.x; i < n; i += stride)
    lmax = fmaxf(lmax, fabsf(src[i]));
  __shared__ float red[256];
  red[threadIdx.x] = lmax;
  __syncthreads();
  for (int s = 128; s > 0; s >>= 1) {
    if (threadIdx.x < (unsigned)s)
      red[threadIdx.x] = fmaxf(red[threadIdx.x], red[threadIdx.x + s]);
    __syncthreads();
  }
  if (threadIdx.x == 0) atomicMax(out, __float_as_uint(red[0]));
}

__global__ __launch_bounds__(256) void quant_pad_kernel(const float* __restrict__ src,
                                                        int8_t* __restrict__ dst,
                                                        int R, int C, int Rp, int Cp,
                                                        const unsigned* amax) {
  float s = q_scale(amax);
  float rs = 1.0f / s;
  size_t total = (size_t)Rp * (size_t)Cp;
  size_t stride = (size_t)gridDim.x * blockDim.x;
  for (size_t i = (size_t)blockIdx.x * blockDim.x + threadIdx.x; i < total; i += stride) {
    int r = (int)(i / (size_t)Cp);
    int c = (int)(i % (size_t)Cp);
    float v = (r < R && c < C) ? src[(size_t)r * C + c] : 0.0f;
    float q = fminf(fmaxf(rintf(v * rs), -QMAXF), QMAXF);
    dst[i] = (int8_t)(int)q;
  }
}

template <int NB>
__global__ __launch_bounds__(256) void gemm_iu8_kernel(
    const int8_t* __restrict__ A,
    const int8_t* __restrict__ B0, const int8_t* __restrict__ B1,
    float* __restrict__ C0, float* __restrict__ C1,
    const float* __restrict__ bias0, const float* __restrict__ bias1,
    const unsigned* amaxA, const unsigned* amaxB0, const unsigned* amaxB1,
    int K, int Nreal, int Cstride) {

  __shared__ int8_t lA[2][128 * LDPITCH];
  __shared__ int8_t lB[2][NB][64 * LDPITCH];

  const int tid    = threadIdx.x;
  const int lane   = tid & 31;
  const int lr     = lane & 15;
  const int lh     = lane >> 4;
  const int waveId = tid >> 5;
  const int waveM  = waveId >> 1;
  const int waveN  = waveId & 1;

  const int blockN = blockIdx.x * 64;
  const int blockM = blockIdx.y * 128;

  auto stage = [&](int buf, int k0) {
#pragma unroll
    for (int i = 0; i < 2; ++i) {
      int c   = tid + i * 256;
      int row = c >> 2;
      int cp  = (c & 3) * 16;
      async_ld_b128(&lA[buf][row * LDPITCH + cp],
                    A + (size_t)(blockM + row) * K + k0 + cp);
    }
    int row = tid >> 2;
    int cp  = (tid & 3) * 16;
    async_ld_b128(&lB[buf][0][row * LDPITCH + cp],
                  B0 + (size_t)(blockN + row) * K + k0 + cp);
    if constexpr (NB == 2)
      async_ld_b128(&lB[buf][1][row * LDPITCH + cp],
                    B1 + (size_t)(blockN + row) * K + k0 + cp);
  };

  v8i acc[NB][2][2];
#pragma unroll
  for (int i = 0; i < NB; ++i)
#pragma unroll
    for (int m = 0; m < 2; ++m)
#pragma unroll
      for (int n = 0; n < 2; ++n)
        acc[i][m][n] = (v8i){0, 0, 0, 0, 0, 0, 0, 0};

  const int nsteps = K >> 6;
  stage(0, 0);

  for (int s = 0; s < nsteps; ++s) {
    const int  cur      = s & 1;
    const bool has_next = (s + 1) < nsteps;

    if (has_next) {
      stage(cur ^ 1, (s + 1) << 6);
      if constexpr (NB == 2) asm volatile("s_wait_asynccnt 0x4" ::: "memory");
      else                   asm volatile("s_wait_asynccnt 0x3" ::: "memory");
      if (s + 2 < nsteps)
        __builtin_prefetch(A + (size_t)(blockM + (tid >> 2)) * K +
                               ((s + 2) << 6) + (tid & 3) * 16, 0, 1);
    } else {
      asm volatile("s_wait_asynccnt 0x0" ::: "memory");
    }
    __syncthreads();

    v8i afrag[2];
#pragma unroll
    for (int m = 0; m < 2; ++m) {
      const int8_t* base = &lA[cur][(waveM * 32 + m * 16 + lr) * LDPITCH];
#pragma unroll
      for (int v = 0; v < 8; ++v) {
        int ko = ((v >> 1) << 4) + ((v & 1) << 2) + (lh << 3);
        afrag[m][v] = *reinterpret_cast<const int*>(base + ko);
      }
    }
    v8i bfrag[NB][2];
#pragma unroll
    for (int i = 0; i < NB; ++i)
#pragma unroll
      for (int n = 0; n < 2; ++n) {
        const int8_t* base = &lB[cur][i][(waveN * 32 + n * 16 + lr) * LDPITCH];
#pragma unroll
        for (int v = 0; v < 8; ++v) {
          int ko = ((v >> 1) << 4) + ((v & 1) << 2) + (lh << 3);
          bfrag[i][n][v] = *reinterpret_cast<const int*>(base + ko);
        }
      }

#pragma unroll
    for (int i = 0; i < NB; ++i)
#pragma unroll
      for (int m = 0; m < 2; ++m)
#pragma unroll
        for (int n = 0; n < 2; ++n)
          acc[i][m][n] = __builtin_amdgcn_wmma_i32_16x16x64_iu8(
              true, afrag[m], true, bfrag[i][n], acc[i][m][n], false, false);

    __syncthreads();
  }

  const float sA = q_scale(amaxA);
  float sB[2];
  sB[0] = q_scale(amaxB0);
  sB[1] = (NB == 2) ? q_scale(amaxB1) : 0.0f;

#pragma unroll
  for (int i = 0; i < NB; ++i) {
    float*       C    = (i == 0) ? C0 : C1;
    const float* bias = (i == 0) ? bias0 : bias1;
    const float  sc   = sA * sB[i];
#pragma unroll
    for (int m = 0; m < 2; ++m)
#pragma unroll
      for (int n = 0; n < 2; ++n) {
        int  col = blockN + waveN * 32 + n * 16 + lr;
        bool ok  = col < Nreal;
        float bv = ok ? bias[col] : 0.0f;
#pragma unroll
        for (int r = 0; r < 8; ++r) {
          int row = blockM + waveM * 32 + m * 16 + r + 8 * lh;
          float v = ok ? fmaf((float)acc[i][m][n][r], sc, bv) : 0.0f;
          C[(size_t)row * Cstride + col] = v;
        }
      }
  }
}

__global__ __launch_bounds__(256) void swiglu_kernel(float* __restrict__ g,
                                                     const float* __restrict__ u,
                                                     size_t n,
                                                     const unsigned* amg,
                                                     const unsigned* amu,
                                                     unsigned* amact) {
  const float sg = q_scale(amg);
  const float su = q_scale(amu);
  float lmax = 0.0f;
  size_t stride = (size_t)gridDim.x * blockDim.x;
  for (size_t i = (size_t)blockIdx.x * blockDim.x + threadIdx.x; i < n; i += stride) {
    float gq = fq_val(g[i], sg);
    float uq = fq_val(u[i], su);
    float a  = (gq / (1.0f + __expf(-gq))) * uq;
    g[i] = a;
    lmax = fmaxf(lmax, fabsf(a));
  }
  __shared__ float red[256];
  red[threadIdx.x] = lmax;
  __syncthreads();
  for (int s = 128; s > 0; s >>= 1) {
    if (threadIdx.x < (unsigned)s)
      red[threadIdx.x] = fmaxf(red[threadIdx.x], red[threadIdx.x + s]);
    __syncthreads();
  }
  if (threadIdx.x == 0) atomicMax(amact, __float_as_uint(red[0]));
}

extern "C" void kernel_launch(void* const* d_in, const int* in_sizes, int n_in,
                              void* d_out, int out_size, void* d_ws, size_t ws_size,
                              hipStream_t stream) {
  const float* x  = (const float*)d_in[0];
  const float* wg = (const float*)d_in[1];
  const float* wu = (const float*)d_in[2];
  const float* wd = (const float*)d_in[3];
  const float* bg = (const float*)d_in[4];
  const float* bu = (const float*)d_in[5];
  const float* bd = (const float*)d_in[6];
  float* out = (float*)d_out;

  char* ws = (char*)d_ws;
  unsigned* slots = (unsigned*)ws;
  size_t off = 256;
  auto carve = [&](size_t bytes) -> char* {
    char* p = ws + off;
    off = (off + bytes + 255) & ~(size_t)255;
    return p;
  };
  int8_t* XQ   = (int8_t*)carve((size_t)T_TOK * D_MODEL);
  int8_t* WGQ  = (int8_t*)carve((size_t)HPAD * D_MODEL);
  int8_t* WUQ  = (int8_t*)carve((size_t)HPAD * D_MODEL);
  int8_t* WDQ  = (int8_t*)carve((size_t)D_MODEL * HPAD);
  float*  GATE = (float*)carve((size_t)T_TOK * HPAD * 4);
  float*  UP   = (float*)carve((size_t)T_TOK * HPAD * 4);
  int8_t* ACT  = (int8_t*)carve((size_t)T_TOK * HPAD);
  (void)ws_size; (void)in_sizes; (void)n_in; (void)out_size;

  init_slots_kernel<<<1, 32, 0, stream>>>(slots);

  absmax_kernel<<<2048, 256, 0, stream>>>(x,  (size_t)T_TOK * D_MODEL,  slots + 0);
  absmax_kernel<<<2048, 256, 0, stream>>>(wg, (size_t)HIDDEN * D_MODEL, slots + 1);
  absmax_kernel<<<2048, 256, 0, stream>>>(wu, (size_t)HIDDEN * D_MODEL, slots + 2);
  absmax_kernel<<<2048, 256, 0, stream>>>(wd, (size_t)D_MODEL * HIDDEN, slots + 3);

  quant_pad_kernel<<<4096, 256, 0, stream>>>(x,  XQ,  T_TOK,  D_MODEL, T_TOK,  D_MODEL, slots + 0);
  quant_pad_kernel<<<4096, 256, 0, stream>>>(wg, WGQ, HIDDEN, D_MODEL, HPAD,   D_MODEL, slots + 1);
  quant_pad_kernel<<<4096, 256, 0, stream>>>(wu, WUQ, HIDDEN, D_MODEL, HPAD,   D_MODEL, slots + 2);
  quant_pad_kernel<<<4096, 256, 0, stream>>>(wd, WDQ, D_MODEL, HIDDEN, D_MODEL, HPAD,   slots + 3);

  gemm_iu8_kernel<2><<<dim3(HPAD / 64, T_TOK / 128), 256, 0, stream>>>(
      XQ, WGQ, WUQ, GATE, UP, bg, bu,
      slots + 0, slots + 1, slots + 2,
      D_MODEL, HIDDEN, HPAD);

  absmax_kernel<<<4096, 256, 0, stream>>>(GATE, (size_t)T_TOK * HPAD, slots + 4);
  absmax_kernel<<<4096, 256, 0, stream>>>(UP,   (size_t)T_TOK * HPAD, slots + 5);

  swiglu_kernel<<<4096, 256, 0, stream>>>(GATE, UP, (size_t)T_TOK * HPAD,
                                          slots + 4, slots + 5, slots + 6);

  quant_pad_kernel<<<4096, 256, 0, stream>>>(GATE, ACT, T_TOK, HPAD, T_TOK, HPAD, slots + 6);

  gemm_iu8_kernel<1><<<dim3(D_MODEL / 64, T_TOK / 128), 256, 0, stream>>>(
      ACT, WDQ, nullptr, out, nullptr, bd, nullptr,
      slots + 6, slots + 3, nullptr,
      HPAD, D_MODEL, D_MODEL);
}
